// _xLSTMEncoder_67912022884856
// MI455X (gfx1250) — compile-verified, matrix operand order checked by reading, not run on hardware
//
#include <hip/hip_runtime.h>

#define T_STEPS 256
#define HDIM    1024
#define EPS_LN  1e-5f
#define SCAN_BLOCKS 64
#define SCAN_THREADS 128

typedef __attribute__((ext_vector_type(16))) __bf16 v16bf;
typedef __attribute__((ext_vector_type(8)))  __bf16 v8bf;
typedef __attribute__((ext_vector_type(8)))  float  v8f;

__device__ __forceinline__ unsigned short f2bf(float f) {
  unsigned u = __float_as_uint(f);
  u += 0x7FFFu + ((u >> 16) & 1u);          // round-to-nearest-even
  return (unsigned short)(u >> 16);
}

__device__ __forceinline__ v16bf load_frag(const unsigned short* p) {
  v8bf lo = *reinterpret_cast<const v8bf*>(p);
  v8bf hi = *reinterpret_cast<const v8bf*>(p + 16);
  return __builtin_shufflevector(lo, hi, 0,1,2,3,4,5,6,7,8,9,10,11,12,13,14,15);
}

#define WMMA_BF16(a, b, c) \
  __builtin_amdgcn_wmma_f32_16x16x32_bf16(false, (a), false, (b), (short)0, (c), false, false)

// Bounded-depth pipelined GEMM over compile-time kt range [KT0, KT1).
// Rolled loop (4 kt / iteration) with explicit P/Q pair ping-pong:
// each pair's 20 b128 loads stay in flight ~1 full iteration ahead of the
// WMMAs that consume them. Final iteration over-reads one pair (pads cover it).
template <int KT0, int KT1>
__device__ __forceinline__ void gemm_region(
    const unsigned short* pA,
    const unsigned short* pB0, const unsigned short* pB1,
    const unsigned short* pB2, const unsigned short* pB3,
    v8f* acc) {
  constexpr int N = KT1 - KT0;
  if constexpr (N <= 0) {
    return;
  } else if constexpr ((N & 3) != 0) {
    // depth-1 fallback for odd shapes (unused by current instantiations)
    v16bf a0  = load_frag(pA  + (KT0 << 5));
    v16bf b00 = load_frag(pB0 + (KT0 << 5));
    v16bf b01 = load_frag(pB1 + (KT0 << 5));
    v16bf b02 = load_frag(pB2 + (KT0 << 5));
    v16bf b03 = load_frag(pB3 + (KT0 << 5));
#pragma unroll
    for (int i = 1; i < N; ++i) {
      const int kt = KT0 + i;
      v16bf a1 = load_frag(pA  + (kt << 5));
      v16bf c0 = load_frag(pB0 + (kt << 5));
      v16bf c1 = load_frag(pB1 + (kt << 5));
      v16bf c2 = load_frag(pB2 + (kt << 5));
      v16bf c3 = load_frag(pB3 + (kt << 5));
      acc[0] = WMMA_BF16(a0, b00, acc[0]);
      acc[1] = WMMA_BF16(a0, b01, acc[1]);
      acc[2] = WMMA_BF16(a0, b02, acc[2]);
      acc[3] = WMMA_BF16(a0, b03, acc[3]);
      a0 = a1; b00 = c0; b01 = c1; b02 = c2; b03 = c3;
    }
    acc[0] = WMMA_BF16(a0, b00, acc[0]);
    acc[1] = WMMA_BF16(a0, b01, acc[1]);
    acc[2] = WMMA_BF16(a0, b02, acc[2]);
    acc[3] = WMMA_BF16(a0, b03, acc[3]);
  } else {
    const unsigned short* a  = pA  + (KT0 << 5);
    const unsigned short* w0 = pB0 + (KT0 << 5);
    const unsigned short* w1 = pB1 + (KT0 << 5);
    const unsigned short* w2 = pB2 + (KT0 << 5);
    const unsigned short* w3 = pB3 + (KT0 << 5);
    // preload pair P = kt {0,1}
    v16bf Ap0 = load_frag(a);        v16bf Ap1 = load_frag(a + 32);
    v16bf P00 = load_frag(w0);       v16bf P01 = load_frag(w0 + 32);
    v16bf P10 = load_frag(w1);       v16bf P11 = load_frag(w1 + 32);
    v16bf P20 = load_frag(w2);       v16bf P21 = load_frag(w2 + 32);
    v16bf P30 = load_frag(w3);       v16bf P31 = load_frag(w3 + 32);
#pragma unroll 1
    for (int i = 0; i < N / 4; ++i) {
      // prefetch pair Q = kt {+2,+3}
      v16bf Aq0 = load_frag(a + 64);   v16bf Aq1 = load_frag(a + 96);
      v16bf Q00 = load_frag(w0 + 64);  v16bf Q01 = load_frag(w0 + 96);
      v16bf Q10 = load_frag(w1 + 64);  v16bf Q11 = load_frag(w1 + 96);
      v16bf Q20 = load_frag(w2 + 64);  v16bf Q21 = load_frag(w2 + 96);
      v16bf Q30 = load_frag(w3 + 64);  v16bf Q31 = load_frag(w3 + 96);
      acc[0] = WMMA_BF16(Ap0, P00, acc[0]);
      acc[1] = WMMA_BF16(Ap0, P10, acc[1]);
      acc[2] = WMMA_BF16(Ap0, P20, acc[2]);
      acc[3] = WMMA_BF16(Ap0, P30, acc[3]);
      acc[0] = WMMA_BF16(Ap1, P01, acc[0]);
      acc[1] = WMMA_BF16(Ap1, P11, acc[1]);
      acc[2] = WMMA_BF16(Ap1, P21, acc[2]);
      acc[3] = WMMA_BF16(Ap1, P31, acc[3]);
      // prefetch next P = kt {+4,+5} (last iteration over-reads into pads)
      Ap0 = load_frag(a + 128);  Ap1 = load_frag(a + 160);
      P00 = load_frag(w0 + 128); P01 = load_frag(w0 + 160);
      P10 = load_frag(w1 + 128); P11 = load_frag(w1 + 160);
      P20 = load_frag(w2 + 128); P21 = load_frag(w2 + 160);
      P30 = load_frag(w3 + 128); P31 = load_frag(w3 + 160);
      acc[0] = WMMA_BF16(Aq0, Q00, acc[0]);
      acc[1] = WMMA_BF16(Aq0, Q10, acc[1]);
      acc[2] = WMMA_BF16(Aq0, Q20, acc[2]);
      acc[3] = WMMA_BF16(Aq0, Q30, acc[3]);
      acc[0] = WMMA_BF16(Aq1, Q01, acc[0]);
      acc[1] = WMMA_BF16(Aq1, Q11, acc[1]);
      acc[2] = WMMA_BF16(Aq1, Q21, acc[2]);
      acc[3] = WMMA_BF16(Aq1, Q31, acc[3]);
      a += 128; w0 += 128; w1 += 128; w2 += 128; w3 += 128;
    }
  }
}

// grid-wide barrier: monotonically increasing epoch on a single counter
__device__ __forceinline__ void gbar(unsigned* bar, unsigned* epoch) {
  __syncthreads();
  __threadfence();
  ++(*epoch);
  if (threadIdx.x == 0) {
    __hip_atomic_fetch_add(bar, 1u, __ATOMIC_RELEASE, __HIP_MEMORY_SCOPE_AGENT);
    const unsigned target = (*epoch) * gridDim.x;
    while (__hip_atomic_load(bar, __ATOMIC_ACQUIRE, __HIP_MEMORY_SCOPE_AGENT) < target)
      __builtin_amdgcn_s_sleep(2);
  }
  __threadfence();
  __syncthreads();
}

// -------- weight transpose + fp32->bf16 : W[K][N] -> Wt[N][K] --------
__global__ void transpose_f32_bf16(const float* __restrict__ W,
                                   unsigned short* __restrict__ Wt,
                                   int K, int N) {
  __shared__ unsigned short tile[32][33];
  const int n0 = blockIdx.x * 32;
  const int k0 = blockIdx.y * 32;
  const int tx = threadIdx.x & 31;
  const int ty = threadIdx.x >> 5;          // 0..7
#pragma unroll
  for (int i = 0; i < 4; ++i) {
    const int k = k0 + ty + i * 8;
    tile[ty + i * 8][tx] = f2bf(W[(size_t)k * N + (n0 + tx)]);
  }
  __syncthreads();
#pragma unroll
  for (int i = 0; i < 4; ++i) {
    const int n = n0 + ty + i * 8;
    Wt[(size_t)n * K + (k0 + tx)] = tile[tx][ty + i * 8];
  }
}

// -------- misc init: series->bf16, zero h buffers / accumulators / barriers ----
__global__ void prep_misc(const float* __restrict__ series, unsigned short* __restrict__ Xbf,
                          unsigned short* __restrict__ hbuf0, unsigned short* __restrict__ hbuf1,
                          float* __restrict__ acc0, float* __restrict__ acc1,
                          unsigned* __restrict__ bar0, unsigned* __restrict__ bar1) {
  const size_t NX   = (size_t)32 * 256 * 256;
  const size_t NHB  = (size_t)2 * 32 * 1024;
  const size_t NACC = 4 * 64;
  const size_t total = NX + 2 * NHB + 2 * NACC + 2;
  for (size_t i = (size_t)blockIdx.x * blockDim.x + threadIdx.x; i < total;
       i += (size_t)gridDim.x * blockDim.x) {
    size_t j = i;
    if (j < NX)        { Xbf[j] = f2bf(series[j]); continue; }
    j -= NX;
    if (j < NHB)       { hbuf0[j] = 0;  continue; }
    j -= NHB;
    if (j < NHB)       { hbuf1[j] = 0;  continue; }
    j -= NHB;
    if (j < NACC)      { acc0[j] = 0.0f; continue; }
    j -= NACC;
    if (j < NACC)      { acc1[j] = 0.0f; continue; }
    j -= NACC;
    if (j == 0) *bar0 = 0u;
    if (j == 1) *bar1 = 0u;
  }
}

// -------- persistent recurrent scan for one layer --------------------------
// 128 output tiles (2 m-tiles x 64 col-groups), each split-K across 2 waves.
// 64 blocks x 4 waves: wave = {tileInBlk = wv&1, khalf = wv>>1}.
template <int K, int Kx, int XMODE>
__global__ __launch_bounds__(SCAN_THREADS, 1)
void xlstm_scan(const unsigned short* __restrict__ Xsrc,
                const unsigned short* __restrict__ Wt,
                const float* __restrict__ bias,
                const float* __restrict__ gamma, const float* __restrict__ beta,
                unsigned short* __restrict__ hbuf,      // [2][32][1024]
                unsigned short* __restrict__ Hout,      // [T][32][1024] or null
                float* __restrict__ out,                // [32][1024] or null
                float* __restrict__ accum,              // [4][2][32]
                unsigned* __restrict__ bar) {
  __shared__ float lds_red[64];                         // [sum|sq][32 rows]
  __shared__ float lds_part[2 * 32 * 33];               // split-K partials, padded
  const int lane  = threadIdx.x & 31;
  const int wv    = threadIdx.x >> 5;
  const int tileInBlk = wv & 1;
  const int khalf     = wv >> 1;                        // which K half
  const int tileId = blockIdx.x * 2 + tileInBlk;        // 0..127
  const int m0  = (tileId & 1) * 16;
  const int j0  = (tileId >> 1) * 16;
  const int lhalf = lane >> 4;
  const int l15   = lane & 15;
  const int colLane = j0 + l15;                         // hidden col
  const int mA    = m0 + l15;                           // batch row for A frag

  constexpr int KT  = K >> 5;
  constexpr int xKT = Kx >> 5;
  constexpr int KH  = KT >> 1;                          // per-wave K half
  constexpr int XE0 = (xKT < KH) ? xKT : KH;            // khalf0: x [0,XE0), h [XE0,KH)
  constexpr int HS1 = (xKT > KH) ? xKT : KH;            // khalf1: x [KH,HS1), h [HS1,KT)

  // Hoisted base pointers: fragment address = base + kt*32 elements.
  const unsigned short* pB0 = Wt + (size_t)(0 * HDIM + colLane) * K + (lhalf << 3);
  const unsigned short* pB1 = Wt + (size_t)(1 * HDIM + colLane) * K + (lhalf << 3);
  const unsigned short* pB2 = Wt + (size_t)(2 * HDIM + colLane) * K + (lhalf << 3);
  const unsigned short* pB3 = Wt + (size_t)(3 * HDIM + colLane) * K + (lhalf << 3);
  const unsigned short* hABase = hbuf + (size_t)mA * HDIM + (lhalf << 3) - Kx;
  const unsigned short* xbase = (XMODE == 0)
      ? Xsrc + (size_t)mA * T_STEPS * 256 + (lhalf << 3)
      : Xsrc + (size_t)mA * HDIM + (lhalf << 3);
  constexpr size_t xstep = (XMODE == 0) ? 256 : (size_t)32 * HDIM;

  float bval[4];
#pragma unroll
  for (int g = 0; g < 4; ++g) bval[g] = bias[g * HDIM + colLane];
  const float gval = gamma[colLane];
  const float betv = beta[colLane];

  v8f cacc;
#pragma unroll
  for (int i = 0; i < 8; ++i) cacc[i] = 0.0f;

  unsigned epoch = 0;

  for (int t = 0; t < T_STEPS; ++t) {
    if (threadIdx.x < 64) lds_red[threadIdx.x] = 0.0f;
    __syncthreads();

    v8f acc[4];                                         // zero C: inline-foldable
#pragma unroll
    for (int g = 0; g < 4; ++g)
#pragma unroll
      for (int i = 0; i < 8; ++i) acc[g][i] = 0.0f;

    const unsigned short* pAx = xbase + (size_t)t * xstep;
    const unsigned short* pAh = hABase + (size_t)(t & 1) * (32 * HDIM);
    if (khalf == 0) {                                   // wave-uniform branch
      gemm_region<0,   XE0>(pAx, pB0, pB1, pB2, pB3, acc);
      gemm_region<XE0, KH >(pAh, pB0, pB1, pB2, pB3, acc);
    } else {
      gemm_region<KH,  HS1>(pAx, pB0, pB1, pB2, pB3, acc);
      gemm_region<HS1, KT >(pAh, pB0, pB1, pB2, pB3, acc);
    }

    // combine split-K partials through LDS (pad 33 -> conflict-free)
    const int pbase = tileInBlk * (32 * 33) + lane * 33;
    if (khalf == 1) {
#pragma unroll
      for (int g = 0; g < 4; ++g)
#pragma unroll
        for (int r = 0; r < 8; ++r)
          lds_part[pbase + g * 8 + r] = acc[g][r];
    }
    __syncthreads();

    if (khalf == 0) {
#pragma unroll
      for (int g = 0; g < 4; ++g)
#pragma unroll
        for (int r = 0; r < 8; ++r)
          acc[g][r] += lds_part[pbase + g * 8 + r] + bval[g];   // bias added here

      // gates + cell update + per-row partial sums
#pragma unroll
      for (int r = 0; r < 8; ++r) {
        const float zi = acc[0][r], zf = acc[1][r], zc = acc[2][r];
        const float ig = __expf(fminf(fmaxf(zi, -6.0f), 3.0f));
        const float fg = __expf(fminf(fmaxf(zf, -6.0f), 3.0f));
        const float cd = tanhf(zc);
        const float c  = fg * cacc[r] + ig * cd;
        cacc[r] = c;
        float s = c, q = c * c;
#pragma unroll
        for (int mk = 1; mk <= 8; mk <<= 1) {
          s += __shfl_xor(s, mk, 32);
          q += __shfl_xor(q, mk, 32);
        }
        if (l15 == 0) {
          const int row = m0 + r + 8 * lhalf;
          atomicAdd(&lds_red[row], s);
          atomicAdd(&lds_red[32 + row], q);
        }
      }
    }
    __syncthreads();

    if (threadIdx.x < 32) {
      const int row = threadIdx.x;
      atomicAdd(&accum[(t & 3) * 64 + row],      lds_red[row]);
      atomicAdd(&accum[(t & 3) * 64 + 32 + row], lds_red[32 + row]);
    }
    if (blockIdx.x == 0 && threadIdx.x >= 64) {         // pre-zero slot t+2
      __hip_atomic_store(&accum[((t + 2) & 3) * 64 + (threadIdx.x - 64)], 0.0f,
                         __ATOMIC_RELAXED, __HIP_MEMORY_SCOPE_AGENT);
    }
    gbar(bar, &epoch);                                  // all row sums visible

    if (khalf == 0) {
      unsigned short* hn = hbuf + (size_t)((t + 1) & 1) * (32 * HDIM);
#pragma unroll
      for (int r = 0; r < 8; ++r) {
        const int row = m0 + r + 8 * lhalf;
        const float sum = __hip_atomic_load(&accum[(t & 3) * 64 + row],
                                            __ATOMIC_RELAXED, __HIP_MEMORY_SCOPE_AGENT);
        const float sq  = __hip_atomic_load(&accum[(t & 3) * 64 + 32 + row],
                                            __ATOMIC_RELAXED, __HIP_MEMORY_SCOPE_AGENT);
        const float mu  = sum * (1.0f / 1024.0f);
        const float var = sq * (1.0f / 1024.0f) - mu * mu;
        const float nrm = (cacc[r] - mu) * rsqrtf(var + EPS_LN) * gval + betv;
        const float sg  = 1.0f / (1.0f + __expf(-acc[3][r]));
        const float h   = sg * tanhf(nrm);
        const unsigned short hb16 = f2bf(h);
        hn[(size_t)row * HDIM + colLane] = hb16;
        if (Hout) Hout[((size_t)t * 32 + row) * HDIM + colLane] = hb16;
        if (out && t == T_STEPS - 1) out[(size_t)row * HDIM + colLane] = h;
      }
    }
    gbar(bar, &epoch);                                  // h visible for step t+1
  }
}

extern "C" void kernel_launch(void* const* d_in, const int* in_sizes, int n_in,
                              void* d_out, int out_size, void* d_ws, size_t ws_size,
                              hipStream_t stream) {
  (void)in_sizes; (void)n_in; (void)out_size; (void)ws_size;
  const float* series = (const float*)d_in[0];
  const float* W0 = (const float*)d_in[1];
  const float* b0 = (const float*)d_in[2];
  const float* g0 = (const float*)d_in[3];
  const float* be0 = (const float*)d_in[4];
  const float* W1 = (const float*)d_in[5];
  const float* b1 = (const float*)d_in[6];
  const float* g1 = (const float*)d_in[7];
  const float* be1 = (const float*)d_in[8];

  char* ws = (char*)d_ws;
  size_t o = 0;                                         // 512 B pads: over-read safety
  unsigned short* Wt0   = (unsigned short*)(ws + o); o += (size_t)4096 * 1280 * 2 + 512;
  unsigned short* Wt1   = (unsigned short*)(ws + o); o += (size_t)4096 * 2048 * 2 + 512;
  unsigned short* Xbf   = (unsigned short*)(ws + o); o += (size_t)32 * 256 * 256 * 2 + 512;
  unsigned short* H0out = (unsigned short*)(ws + o); o += (size_t)256 * 32 * 1024 * 2 + 512;
  unsigned short* hbuf0 = (unsigned short*)(ws + o); o += (size_t)2 * 32 * 1024 * 2 + 512;
  unsigned short* hbuf1 = (unsigned short*)(ws + o); o += (size_t)2 * 32 * 1024 * 2 + 512;
  float*    acc0 = (float*)(ws + o);    o += 4 * 64 * sizeof(float) + 512;
  float*    acc1 = (float*)(ws + o);    o += 4 * 64 * sizeof(float) + 512;
  unsigned* bar0 = (unsigned*)(ws + o); o += 256;
  unsigned* bar1 = (unsigned*)(ws + o); o += 256;      // total ~48.6 MB

  transpose_f32_bf16<<<dim3(4096 / 32, 1280 / 32), 256, 0, stream>>>(W0, Wt0, 1280, 4096);
  transpose_f32_bf16<<<dim3(4096 / 32, 2048 / 32), 256, 0, stream>>>(W1, Wt1, 2048, 4096);
  prep_misc<<<2048, 256, 0, stream>>>(series, Xbf, hbuf0, hbuf1, acc0, acc1, bar0, bar1);

  // layer-0: K = 256 (x) + 1024 (h)
  xlstm_scan<1280, 256, 0><<<SCAN_BLOCKS, SCAN_THREADS, 0, stream>>>(
      Xbf, Wt0, b0, g0, be0, hbuf0, H0out, nullptr, acc0, bar0);

  // layer-1: K = 1024 (layer-0 h) + 1024 (h); emits last h fp32
  xlstm_scan<2048, 1024, 1><<<SCAN_BLOCKS, SCAN_THREADS, 0, stream>>>(
      H0out, Wt1, b1, g1, be1, hbuf1, nullptr, (float*)d_out, acc1, bar1);
}
